// PoseMixtureVAE_37477884625416
// MI455X (gfx1250) — matrix operand order repaired and checked by reading, not compiled, not run on hardware
//
#include <hip/hip_runtime.h>
#include <hip/hip_bf16.h>

typedef _Float16 v16h __attribute__((ext_vector_type(16)));
typedef _Float16 v8h  __attribute__((ext_vector_type(8)));
typedef float    v8f  __attribute__((ext_vector_type(8)));
typedef int      v4i  __attribute__((vector_size(4 * sizeof(int))));

#define BDIM 8192
#define FDIM 267
#define LDIM 32
#define HDIM 1024
#define EDIM 8
#define GDIM 64

#if defined(__has_builtin)
#  if __has_builtin(__builtin_amdgcn_global_load_async_to_lds_b128)
#    define HAVE_ASYNC_B128 1
#  endif
#  if __has_builtin(__builtin_amdgcn_s_wait_asynccnt)
#    define HAVE_WAIT_ASYNC 1
#  endif
#endif

typedef __attribute__((address_space(1))) v4i* gv4i_p;   // global v4i*
typedef __attribute__((address_space(3))) v4i* lv4i_p;   // LDS v4i*

// 16-byte global -> LDS async copy (ASYNCcnt-tracked, GLOBAL_LOAD_ASYNC_TO_LDS_B128)
__device__ __forceinline__ void async_cp_b128(_Float16* l, const _Float16* g)
{
#if defined(HAVE_ASYNC_B128)
    __builtin_amdgcn_global_load_async_to_lds_b128(
        (gv4i_p)(unsigned long long)g,
        (lv4i_p)(unsigned)(unsigned long long)l, 0, 0);
#else
    unsigned laddr = (unsigned)(unsigned long long)l;
    asm volatile("global_load_async_to_lds_b128 %0, %1, off"
                 :: "v"(laddr), "v"(g) : "memory");
#endif
}

template <int N>
__device__ __forceinline__ void wait_async()
{
#if defined(HAVE_WAIT_ASYNC)
    __builtin_amdgcn_s_wait_asynccnt(N);
#else
    asm volatile("s_wait_asynccnt %0" :: "i"(N) : "memory");
#endif
}

__device__ __forceinline__ float elu_f(float x) { return x > 0.f ? x : __expf(x) - 1.f; }

// ---------------------------------------------------------------------------
// Tiled f16 WMMA GEMM with async-LDS double buffering.
//   C[B x Ntrue] (f32) = A[B x K] (f16, row-major) * Wt (f16, TRANSPOSED: [N][K])
// Block tile 128x128, 8 waves (4 M x 2 N), each wave 2x4 v_wmma_f32_16x16x32_f16.
// Expert mode (coef != null): A is inp[B x inpad] reused per expert
// (k = e*inpad + i); coef[b,e] is applied as an accumulator fold at expert
// boundaries, so staging is a pure async copy in both modes.
// ---------------------------------------------------------------------------
__global__ __launch_bounds__(256)
void gemm_wmma_f16(const _Float16* __restrict__ A, int ldA,
                   const _Float16* __restrict__ Wt, int ldW /* == Kp */,
                   float* __restrict__ C, int ldC, int Ntrue,
                   int Kp,
                   const float* __restrict__ coef, int inpad)
{
    __shared__ alignas(16) _Float16 At[2][128 * 32];   // [row][k]
    __shared__ alignas(16) _Float16 Bt[2][128 * 32];   // [n][k]

    const int tid   = threadIdx.x;
    const int bm    = blockIdx.y;
    const int bn    = blockIdx.x;
    const int lane  = tid & 31;
    const int wv    = tid >> 5;
    const int waveM = wv & 3;            // 0..3 -> M offset *32
    const int waveN = wv >> 2;           // 0..1 -> N offset *64
    const int laneHalf = lane >> 4;
    const int laneM    = lane & 15;

    // staging: each thread copies 16 contiguous halves (2 x b128) of A and of B
    const int srow = tid >> 1;           // 0..127 (A row / B n-row)
    const int skb  = (tid & 1) << 4;     // 0 / 16 halves

    const int  inpadC  = coef ? inpad : Kp;
    const long strideA = coef ? inpad : ldA;
    const long grow    = (long)bm * 128 + srow;          // A source row
    const long brow    = (long)bn * 128 + srow;          // Wt source row (n)

    v8f acc[2][4], pacc[2][4];
#pragma unroll
    for (int mi = 0; mi < 2; ++mi)
#pragma unroll
        for (int ni = 0; ni < 4; ++ni)
#pragma unroll
            for (int r = 0; r < 8; ++r) { acc[mi][ni][r] = 0.f; pacc[mi][ni][r] = 0.f; }

    const int nIter = Kp >> 5;

    // uniform stage counters (division-free expert tracking)
    int s_kin = 0, s_e = 0;
    auto issue = [&](int buf, int kin, int e) {
        const _Float16* aSrc = A + grow * strideA + kin + skb;
        _Float16*       aDst = &At[buf][srow * 32 + skb];
        async_cp_b128(aDst, aSrc);
        async_cp_b128(aDst + 8, aSrc + 8);
        const int kt = e * inpadC + kin;                 // global K for weights
        const _Float16* bSrc = Wt + brow * ldW + kt + skb;
        _Float16*       bDst = &Bt[buf][srow * 32 + skb];
        async_cp_b128(bDst, bSrc);
        async_cp_b128(bDst + 8, bSrc + 8);
    };
    auto advance = [&]() { s_kin += 32; if (s_kin == inpadC) { s_kin = 0; ++s_e; } };

    issue(0, s_kin, s_e);
    advance();

    int kin = 0, e = 0;
    for (int i = 0; i < nIter; ++i) {
        const int cur = i & 1;
        if (i + 1 < nIter) { issue(cur ^ 1, s_kin, s_e); advance(); wait_async<4>(); }
        else               { wait_async<0>(); }
        __syncthreads();   // all waves' tiles for buf[cur] resident

        // ---- fragments (documented 16-bit A/B VGPR layouts) ----
        v16h af[2], bf[4];
#pragma unroll
        for (int mi = 0; mi < 2; ++mi) {
            const _Float16* p = &At[cur][(waveM * 32 + mi * 16 + laneM) * 32 + laneHalf * 8];
            v8h lo = *(const v8h*)p;            // K = half*8 .. +7
            v8h hi = *(const v8h*)(p + 16);     // K = 16+half*8 .. +7
#pragma unroll
            for (int q = 0; q < 8; ++q) { af[mi][q] = lo[q]; af[mi][q + 8] = hi[q]; }
        }
#pragma unroll
        for (int ni = 0; ni < 4; ++ni) {
            const _Float16* p = &Bt[cur][(waveN * 64 + ni * 16 + laneM) * 32 + laneHalf * 8];
            v8h lo = *(const v8h*)p;            // K = half*8 .. +7
            v8h hi = *(const v8h*)(p + 16);     // K = 16+half*8 .. +7
#pragma unroll
            for (int q = 0; q < 8; ++q) { bf[ni][q] = lo[q]; bf[ni][q + 8] = hi[q]; }
        }

#pragma unroll
        for (int mi = 0; mi < 2; ++mi)
#pragma unroll
            for (int ni = 0; ni < 4; ++ni)
                pacc[mi][ni] = __builtin_amdgcn_wmma_f32_16x16x32_f16(
                    false, af[mi], false, bf[ni], (short)0, pacc[mi][ni], false, false);

        kin += 32;
        if (kin == inpadC) {               // expert boundary: fold partial into acc
            if (coef) {                    // uniform branch; unconditional loads inside
#pragma unroll
                for (int mi = 0; mi < 2; ++mi) {
                    const int row0 = bm * 128 + waveM * 32 + mi * 16 + laneHalf * 8;
                    float cf[8];
#pragma unroll
                    for (int r = 0; r < 8; ++r)
                        cf[r] = coef[(long)(row0 + r) * EDIM + e];
#pragma unroll
                    for (int ni = 0; ni < 4; ++ni)
#pragma unroll
                        for (int r = 0; r < 8; ++r) {
                            acc[mi][ni][r] += cf[r] * pacc[mi][ni][r];
                            pacc[mi][ni][r] = 0.f;
                        }
                }
            } else {
#pragma unroll
                for (int mi = 0; mi < 2; ++mi)
#pragma unroll
                    for (int ni = 0; ni < 4; ++ni)
#pragma unroll
                        for (int r = 0; r < 8; ++r) {
                            acc[mi][ni][r] += pacc[mi][ni][r];
                            pacc[mi][ni][r] = 0.f;
                        }
            }
            kin = 0; ++e;
        }
        __syncthreads();   // all waves done reading buf[cur] before it is refilled
    }

    // ---- store C (16x16 f32 layout: lanes 0-15 -> M=r, lanes 16-31 -> M=8+r) ----
#pragma unroll
    for (int mi = 0; mi < 2; ++mi) {
        const int row0 = bm * 128 + waveM * 32 + mi * 16 + laneHalf * 8;
#pragma unroll
        for (int ni = 0; ni < 4; ++ni) {
            const int col = bn * 128 + waveN * 64 + ni * 16 + laneM;
            if (col < Ntrue) {
#pragma unroll
                for (int r = 0; r < 8; ++r)
                    C[(long)(row0 + r) * ldC + col] = acc[mi][ni][r];
            }
        }
    }
}

// ---------------------------------------------------------------------------
// f32 weight [K x N] -> zero-padded TRANSPOSED f16 pack: dst[(nOff+j)*ldK + kOff + kp]
// ---------------------------------------------------------------------------
__global__ void pack_weight_t(const float* __restrict__ src, int K, int N,
                              _Float16* __restrict__ dst, long ldK, int nOff, int kOff,
                              int Kp, int nSpan)
{
    long idx = (long)blockIdx.x * blockDim.x + threadIdx.x;
    long total = (long)Kp * nSpan;
    if (idx >= total) return;
    int kp = (int)(idx % Kp);
    int j  = (int)(idx / Kp);
    float v = (kp < K && j < N) ? src[(long)kp * N + j] : 0.f;
    dst[(long)(nOff + j) * ldK + kOff + kp] = (_Float16)v;
}

// f32 activation -> f16 pack with zero padding beyond srcN
__global__ void pack_act(const float* __restrict__ src, int srcN,
                         _Float16* __restrict__ dst, int ldDst, int colOff, int span)
{
    long idx = (long)blockIdx.x * blockDim.x + threadIdx.x;
    long total = (long)BDIM * span;
    if (idx >= total) return;
    int  n = (int)(idx % span);
    long b = idx / span;
    float v = (n < srcN) ? src[b * srcN + n] : 0.f;
    dst[b * ldDst + colOff + n] = (_Float16)v;
}

// ---------------------------------------------------------------------------
// Epilogue: bias (plain or expert coef@Bb), optional ELU, write f16 pack
// and/or f32 output. Covers [0, span) columns, zero beyond Ntrue.
// ---------------------------------------------------------------------------
__global__ void epilogue(const float* __restrict__ C, int ldC, int Ntrue, int span,
                         const float* __restrict__ bias,
                         const float* __restrict__ coef, const float* __restrict__ eb, int ebN,
                         int act,
                         _Float16* __restrict__ dstH, int ldH, int colOff,
                         float* __restrict__ dstF, int ldF)
{
    long idx = (long)blockIdx.x * blockDim.x + threadIdx.x;
    long total = (long)BDIM * span;
    if (idx >= total) return;
    int  n = (int)(idx % span);
    long b = idx / span;
    float v = 0.f;
    if (n < Ntrue) {
        v = C[b * ldC + n];
        if (bias) v += bias[n];
        if (coef) {
            const float* cf = coef + b * EDIM;
            float s = 0.f;
#pragma unroll
            for (int ee = 0; ee < EDIM; ++ee) s += cf[ee] * eb[(long)ee * ebN + n];
            v += s;
        }
        if (act) v = elu_f(v);
    }
    if (dstH) dstH[b * ldH + colOff + n] = (_Float16)v;
    if (dstF) dstF[b * ldF + n] = v;
}

// mu / logvar / z reparameterization; writes mu,lv to d_out and f16 z into
// the gate input pack and both decoder input packs (columns 0..31).
__global__ void mu_lv_z(const float* __restrict__ C,   // [B x 64]: mu | lv
                        const float* __restrict__ mu_b, const float* __restrict__ lv_b,
                        const float* __restrict__ eps,
                        float* __restrict__ omu, float* __restrict__ olv,
                        _Float16* __restrict__ gp, int ldG,
                        _Float16* __restrict__ d1, int ld1,
                        _Float16* __restrict__ d2, int ld2)
{
    long idx = (long)blockIdx.x * blockDim.x + threadIdx.x;
    if (idx >= (long)BDIM * LDIM) return;
    int  j = (int)(idx % LDIM);
    long b = idx / LDIM;
    float mu = C[b * 64 + j] + mu_b[j];
    float lv = C[b * 64 + 32 + j] + lv_b[j];
    float z  = mu + eps[b * LDIM + j] * __expf(0.5f * lv);
    omu[b * LDIM + j] = mu;
    olv[b * LDIM + j] = lv;
    _Float16 zh = (_Float16)z;
    gp[b * ldG + j] = zh;
    d1[b * ld1 + j] = zh;
    d2[b * ld2 + j] = zh;
}

__global__ void softmax_e(const float* __restrict__ C, int ldC,
                          const float* __restrict__ b3, float* __restrict__ coef)
{
    long b = (long)blockIdx.x * blockDim.x + threadIdx.x;
    if (b >= BDIM) return;
    float v[EDIM], m = -1e30f;
#pragma unroll
    for (int e = 0; e < EDIM; ++e) { v[e] = C[b * ldC + e] + b3[e]; m = fmaxf(m, v[e]); }
    float s = 0.f;
#pragma unroll
    for (int e = 0; e < EDIM; ++e) { v[e] = __expf(v[e] - m); s += v[e]; }
    float r = 1.f / s;
#pragma unroll
    for (int e = 0; e < EDIM; ++e) coef[b * EDIM + e] = v[e] * r;
}

// ---------------------------------------------------------------------------
extern "C" void kernel_launch(void* const* d_in, const int* in_sizes, int n_in,
                              void* d_out, int out_size, void* d_ws, size_t ws_size,
                              hipStream_t stream)
{
    const float* x      = (const float*)d_in[0];
    const float* c      = (const float*)d_in[1];
    const float* eps    = (const float*)d_in[2];
    const float* enc_w1 = (const float*)d_in[3];
    const float* enc_b1 = (const float*)d_in[4];
    const float* enc_w2 = (const float*)d_in[5];
    const float* enc_b2 = (const float*)d_in[6];
    const float* mu_w   = (const float*)d_in[7];
    const float* mu_b   = (const float*)d_in[8];
    const float* lv_w   = (const float*)d_in[9];
    const float* lv_b   = (const float*)d_in[10];
    const float* gw1    = (const float*)d_in[11];
    const float* gb1    = (const float*)d_in[12];
    const float* gw2    = (const float*)d_in[13];
    const float* gb2    = (const float*)d_in[14];
    const float* gw3    = (const float*)d_in[15];
    const float* gb3    = (const float*)d_in[16];
    const float* w0     = (const float*)d_in[17];
    const float* b0     = (const float*)d_in[18];
    const float* w1     = (const float*)d_in[19];
    const float* b1     = (const float*)d_in[20];
    const float* w2     = (const float*)d_in[21];
    const float* b2     = (const float*)d_in[22];

    float* out0 = (float*)d_out;                       // layer_out [B x F]
    float* omu  = out0 + (long)BDIM * FDIM;            // mu [B x L]
    float* olv  = omu + (long)BDIM * LDIM;             // logvar [B x L]

    // ---- workspace bump allocator ----
    char*  ws  = (char*)d_ws;
    size_t off = 0;
    auto alloc = [&](size_t bytes) -> char* {
        char* p = ws + off;
        off += (bytes + 255) & ~(size_t)255;
        return p;
    };
    float*     Cbuf  = (float*)alloc((size_t)BDIM * HDIM * 4);       // gemm scratch
    _Float16*  a1    = (_Float16*)alloc((size_t)BDIM * 544 * 2);     // [x,c] pad
    _Float16*  a2    = (_Float16*)alloc((size_t)BDIM * 1312 * 2);    // [x,h1] -> [x,h2]
    _Float16*  gpk   = (_Float16*)alloc((size_t)BDIM * 320 * 2);     // [z,c] pad
    _Float16*  g1p   = (_Float16*)alloc((size_t)BDIM * 64 * 2);
    _Float16*  g2p   = (_Float16*)alloc((size_t)BDIM * 64 * 2);
    float*     coefb = (float*)alloc((size_t)BDIM * EDIM * 4);
    _Float16*  din1  = (_Float16*)alloc((size_t)BDIM * 1056 * 2);    // [z, l0]
    _Float16*  din2  = (_Float16*)alloc((size_t)BDIM * 1056 * 2);    // [z, l1]
    // transposed f16 weights [N][K]
    _Float16*  ew1t  = (_Float16*)alloc((size_t)1024 * 544 * 2);
    _Float16*  ew2t  = (_Float16*)alloc((size_t)1024 * 1312 * 2);
    _Float16*  mvt   = (_Float16*)alloc((size_t)128 * 1312 * 2);     // mu|lv|pad
    _Float16*  gw1t  = (_Float16*)alloc((size_t)128 * 320 * 2);
    _Float16*  gw2t  = (_Float16*)alloc((size_t)128 * 64 * 2);
    _Float16*  gw3t  = (_Float16*)alloc((size_t)128 * 64 * 2);
    _Float16*  w0t   = (_Float16*)alloc((size_t)1024 * 2560 * 2);    // [1024][E*320]
    _Float16*  w1t   = (_Float16*)alloc((size_t)1024 * 8448 * 2);    // [1024][E*1056]
    _Float16*  w2t   = (_Float16*)alloc((size_t)384 * 8448 * 2);     // [384][E*1056]

    auto blocks = [](long total) { return dim3((unsigned)((total + 255) / 256)); };

    // ---- weight packs (f32 -> padded, transposed f16) ----
    pack_weight_t<<<blocks((long)544 * 1024), 256, 0, stream>>>(enc_w1, 534, 1024, ew1t, 544, 0, 0, 544, 1024);
    pack_weight_t<<<blocks((long)1312 * 1024), 256, 0, stream>>>(enc_w2, 1291, 1024, ew2t, 1312, 0, 0, 1312, 1024);
    pack_weight_t<<<blocks((long)1312 * 32), 256, 0, stream>>>(mu_w, 1291, 32, mvt, 1312, 0, 0, 1312, 32);
    pack_weight_t<<<blocks((long)1312 * 32), 256, 0, stream>>>(lv_w, 1291, 32, mvt, 1312, 32, 0, 1312, 32);
    pack_weight_t<<<blocks((long)1312 * 64), 256, 0, stream>>>(nullptr, 0, 0, mvt, 1312, 64, 0, 1312, 64);
    pack_weight_t<<<blocks((long)320 * 128), 256, 0, stream>>>(gw1, 299, 64, gw1t, 320, 0, 0, 320, 128);
    pack_weight_t<<<blocks((long)64 * 128), 256, 0, stream>>>(gw2, 64, 64, gw2t, 64, 0, 0, 64, 128);
    pack_weight_t<<<blocks((long)64 * 128), 256, 0, stream>>>(gw3, 64, 8, gw3t, 64, 0, 0, 64, 128);
    for (int e = 0; e < EDIM; ++e) {
        pack_weight_t<<<blocks((long)320 * 1024), 256, 0, stream>>>(
            w0 + (long)e * 299 * 1024, 299, 1024, w0t, 2560, 0, e * 320, 320, 1024);
        pack_weight_t<<<blocks((long)1056 * 1024), 256, 0, stream>>>(
            w1 + (long)e * 1056 * 1024, 1056, 1024, w1t, 8448, 0, e * 1056, 1056, 1024);
        pack_weight_t<<<blocks((long)1056 * 384), 256, 0, stream>>>(
            w2 + (long)e * 1056 * 267, 1056, 267, w2t, 8448, 0, e * 1056, 1056, 384);
    }

    // ---- activation packs ----
    pack_act<<<blocks((long)BDIM * 267), 256, 0, stream>>>(x, 267, a1, 544, 0, 267);
    pack_act<<<blocks((long)BDIM * 277), 256, 0, stream>>>(c, 267, a1, 544, 267, 277);
    pack_act<<<blocks((long)BDIM * 267), 256, 0, stream>>>(x, 267, a2, 1312, 0, 267);

    // ---- encoder layer 1: h1 = elu([x,c] @ enc_w1 + b1) ----
    gemm_wmma_f16<<<dim3(8, 64), 256, 0, stream>>>(a1, 544, ew1t, 544, Cbuf, 1024, 1024, 544, nullptr, 544);
    epilogue<<<blocks((long)BDIM * 1045), 256, 0, stream>>>(Cbuf, 1024, 1024, 1045, enc_b1,
        nullptr, nullptr, 0, 1, a2, 1312, 267, nullptr, 0);

    // ---- encoder layer 2: h2 = elu([x,h1] @ enc_w2 + b2); s = [x,h2] in a2 ----
    gemm_wmma_f16<<<dim3(8, 64), 256, 0, stream>>>(a2, 1312, ew2t, 1312, Cbuf, 1024, 1024, 1312, nullptr, 1312);
    epilogue<<<blocks((long)BDIM * 1045), 256, 0, stream>>>(Cbuf, 1024, 1024, 1045, enc_b2,
        nullptr, nullptr, 0, 1, a2, 1312, 267, nullptr, 0);

    // ---- mu | logvar (fused N=64 GEMM), then z ----
    gemm_wmma_f16<<<dim3(1, 64), 256, 0, stream>>>(a2, 1312, mvt, 1312, Cbuf, 64, 64, 1312, nullptr, 1312);
    mu_lv_z<<<blocks((long)BDIM * LDIM), 256, 0, stream>>>(Cbuf, mu_b, lv_b, eps,
        omu, olv, gpk, 320, din1, 1056, din2, 1056);
    pack_act<<<blocks((long)BDIM * 288), 256, 0, stream>>>(c, 267, gpk, 320, 32, 288);

    // ---- gate MLP + softmax -> coef ----
    gemm_wmma_f16<<<dim3(1, 64), 256, 0, stream>>>(gpk, 320, gw1t, 320, Cbuf, 64, 64, 320, nullptr, 320);
    epilogue<<<blocks((long)BDIM * 64), 256, 0, stream>>>(Cbuf, 64, 64, 64, gb1,
        nullptr, nullptr, 0, 1, g1p, 64, 0, nullptr, 0);
    gemm_wmma_f16<<<dim3(1, 64), 256, 0, stream>>>(g1p, 64, gw2t, 64, Cbuf, 64, 64, 64, nullptr, 64);
    epilogue<<<blocks((long)BDIM * 64), 256, 0, stream>>>(Cbuf, 64, 64, 64, gb2,
        nullptr, nullptr, 0, 1, g2p, 64, 0, nullptr, 0);
    gemm_wmma_f16<<<dim3(1, 64), 256, 0, stream>>>(g2p, 64, gw3t, 64, Cbuf, 8, 8, 64, nullptr, 64);
    softmax_e<<<blocks(BDIM), 256, 0, stream>>>(Cbuf, 8, gb3, coefb);

    // ---- MoE decoder: single-GEMM formulation, coef folded at expert bounds ----
    gemm_wmma_f16<<<dim3(8, 64), 256, 0, stream>>>(gpk, 320, w0t, 2560, Cbuf, 1024, 1024, 2560, coefb, 320);
    epilogue<<<blocks((long)BDIM * 1024), 256, 0, stream>>>(Cbuf, 1024, 1024, 1024, nullptr,
        coefb, b0, 1024, 1, din1, 1056, 32, nullptr, 0);
    gemm_wmma_f16<<<dim3(8, 64), 256, 0, stream>>>(din1, 1056, w1t, 8448, Cbuf, 1024, 1024, 8448, coefb, 1056);
    epilogue<<<blocks((long)BDIM * 1024), 256, 0, stream>>>(Cbuf, 1024, 1024, 1024, nullptr,
        coefb, b1, 1024, 1, din2, 1056, 32, nullptr, 0);
    gemm_wmma_f16<<<dim3(3, 64), 256, 0, stream>>>(din2, 1056, w2t, 8448, Cbuf, 267, 267, 8448, coefb, 1056);
    epilogue<<<blocks((long)BDIM * 267), 256, 0, stream>>>(Cbuf, 267, 267, 267, nullptr,
        coefb, b2, 267, 0, nullptr, 0, 0, out0, 267);
}
